// Attention_CRF_27118423507079
// MI455X (gfx1250) — matrix operand order repaired and checked by reading, not compiled, not run on hardware
//
#include <hip/hip_runtime.h>
#include <math.h>

// ---------------------------------------------------------------------------
// CDNA5 (gfx1250) forward pass for the Attention_CRF reference.
// All heavy matmuls go through v_wmma_f32_16x16x32_f16 (wave32 WMMA).
// LDS tiles are stored fragment-major so per-lane WMMA operands are one
// aligned 32B read (2x ds_load_b128) instead of 16 scalar ds_load_u16.
// ---------------------------------------------------------------------------

typedef __attribute__((ext_vector_type(16))) _Float16 v16h;
typedef __attribute__((ext_vector_type(2)))  _Float16 h2;
typedef __attribute__((ext_vector_type(8)))  float    v8f;

#define BATCH 2048
#define SEQ   70
#define XCOLS 842      // 70 ids + 4 aux + 768 msa
#define EMBD  256
#define DIN   128

#define TPB 256
#define BM  128
#define BN  64
#define BK  32

__device__ __forceinline__ float sigf(float x) { return 1.f / (1.f + __expf(-x)); }

// ---------------------------------------------------------------------------
// Generic WMMA GEMM:  D = act( A*B + bias + Cadd ), f32 in/out, f16 compute.
//   A: M x K (transA: stored K x M), row stride lda, batch stride bsA
//   B: K x N (transB: stored N x K), row stride ldb, batch stride bsB
//   D/Cadd: M x N, row stride ldd, batch stride bsD
// 256 threads = 8 waves; block tile 128x64; each wave owns a 32x32 slice
// (2 A-frags x 2 B-frags -> 4 WMMAs per K-chunk).
//
// Fragment-major LDS layout per 16x16-subtile: [subtile][lane][elem(16)]
//   A (16x32 f16): lane = ((k>>3)&1)*16 + m ; elem = (k&7) + ((k>>4)<<3)
//   B (32x16 f16): lane = ((k>>3)&1)*16 + n ; elem = (k&7) + ((k>>4)<<3)
// so a lane's v16h operand is 16 contiguous halves (32B aligned).
// ---------------------------------------------------------------------------
__launch_bounds__(TPB)
__global__ void gemm_wmma_f16(const float* __restrict__ A,
                              const float* __restrict__ B,
                              const float* __restrict__ bias,
                              const float* __restrict__ Cadd,
                              float* __restrict__ D,
                              int M, int N, int K,
                              int lda, int ldb, int ldd,
                              long long bsA, long long bsB, long long bsD,
                              int transA, int transB, int act)
{
    __shared__ __align__(32) _Float16 AsF[(BM / 16) * 512];   // 4096 halves
    __shared__ __align__(32) _Float16 BsF[(BN / 16) * 512];   // 2048 halves

    const int tid  = threadIdx.x;
    const int wave = tid >> 5;
    const int lane = tid & 31;
    const int lh   = lane >> 4;      // lane half
    const int r    = lane & 15;
    const int wmL  = (wave & 3) * 32;   // wave's M offset in block tile
    const int wnL  = (wave >> 2) * 32;  // wave's N offset in block tile
    const int mtb  = (wave & 3) * 2;    // first A subtile index
    const int ntb  = (wave >> 2) * 2;   // first B subtile index

    const int m0 = blockIdx.y * BM;
    const int n0 = blockIdx.x * BN;
    const long long bA = (long long)blockIdx.z * bsA;
    const long long bB = (long long)blockIdx.z * bsB;
    const long long bD = (long long)blockIdx.z * bsD;

    v8f acc[2][2];
#pragma unroll
    for (int i = 0; i < 2; ++i)
#pragma unroll
        for (int j = 0; j < 2; ++j)
#pragma unroll
            for (int v = 0; v < 8; ++v) acc[i][j][v] = 0.f;

    const int nk = (K + BK - 1) / BK;
    for (int kc = 0; kc < nk; ++kc) {
        const int k0 = kc * BK;

        // --- prefetch next A chunk (global_prefetch_b8, speculative) ---
        if (kc + 1 < nk) {
            int pm = m0 + (tid >> 1);
            int pk = k0 + BK + ((tid & 1) << 4);
            if (pm < M && pk < K) {
                const float* pp = transA ? &A[bA + (long long)pk * lda + pm]
                                         : &A[bA + (long long)pm * lda + pk];
                __builtin_prefetch(pp, 0, 1);
            }
        }

        // --- stage A tile (BM x BK) as f16 pairs, fragment-major ---
#pragma unroll
        for (int i = 0; i < 8; ++i) {
            int e    = tid + i * TPB;     // 0..2047 (pair index)
            int rowR = e >> 4;            // 0..127
            int kk   = (e & 15) << 1;     // even k in 0..30
            int gm   = m0 + rowR;
            int gk   = k0 + kk;
            float v0 = 0.f, v1 = 0.f;
            if (gm < M) {
                if (transA) {
                    if (gk < K)     v0 = A[bA + (long long)gk * lda + gm];
                    if (gk + 1 < K) v1 = A[bA + (long long)(gk + 1) * lda + gm];
                } else {
                    if (gk < K)     v0 = A[bA + (long long)gm * lda + gk];
                    if (gk + 1 < K) v1 = A[bA + (long long)gm * lda + gk + 1];
                }
            }
            int lane_s = (((kk >> 3) & 1) << 4) + (rowR & 15);
            int elem   = (kk & 7) + ((kk >> 4) << 3);
            int addr   = ((rowR >> 4) << 9) + (lane_s << 4) + elem;   // even
            h2 pv; pv[0] = (_Float16)v0; pv[1] = (_Float16)v1;
            *reinterpret_cast<h2*>(&AsF[addr]) = pv;
        }
        // --- stage B tile (BK x BN) as f16 pairs, fragment-major ---
#pragma unroll
        for (int i = 0; i < 4; ++i) {
            int e  = tid + i * TPB;       // 0..1023
            int nn = e >> 4;              // 0..63
            int kk = (e & 15) << 1;       // even k
            int gn = n0 + nn;
            int gk = k0 + kk;
            float v0 = 0.f, v1 = 0.f;
            if (gn < N) {
                if (transB) {
                    if (gk < K)     v0 = B[bB + (long long)gn * ldb + gk];
                    if (gk + 1 < K) v1 = B[bB + (long long)gn * ldb + gk + 1];
                } else {
                    if (gk < K)     v0 = B[bB + (long long)gk * ldb + gn];
                    if (gk + 1 < K) v1 = B[bB + (long long)(gk + 1) * ldb + gn];
                }
            }
            int lane_s = (((kk >> 3) & 1) << 4) + (nn & 15);
            int elem   = (kk & 7) + ((kk >> 4) << 3);
            int addr   = ((nn >> 4) << 9) + (lane_s << 4) + elem;     // even
            h2 pv; pv[0] = (_Float16)v0; pv[1] = (_Float16)v1;
            *reinterpret_cast<h2*>(&BsF[addr]) = pv;
        }
        __syncthreads();

        // --- fragment loads: one aligned 32B read each ---
        const v16h* Af = reinterpret_cast<const v16h*>(AsF);
        const v16h* Bf = reinterpret_cast<const v16h*>(BsF);
        v16h a0 = Af[(mtb + 0) * 32 + lane];
        v16h a1 = Af[(mtb + 1) * 32 + lane];
        v16h b0 = Bf[(ntb + 0) * 32 + lane];
        v16h b1 = Bf[(ntb + 1) * 32 + lane];

        acc[0][0] = __builtin_amdgcn_wmma_f32_16x16x32_f16(
                        false, a0, false, b0, (short)0, acc[0][0], false, false);
        acc[0][1] = __builtin_amdgcn_wmma_f32_16x16x32_f16(
                        false, a0, false, b1, (short)0, acc[0][1], false, false);
        acc[1][0] = __builtin_amdgcn_wmma_f32_16x16x32_f16(
                        false, a1, false, b0, (short)0, acc[1][0], false, false);
        acc[1][1] = __builtin_amdgcn_wmma_f32_16x16x32_f16(
                        false, a1, false, b1, (short)0, acc[1][1], false, false);
        __syncthreads();
    }

    // --- epilogue: D layout lane r = N col, VGPR v = row (v / 8+v per half) ---
#pragma unroll
    for (int mi = 0; mi < 2; ++mi) {
#pragma unroll
        for (int nj = 0; nj < 2; ++nj) {
            int col = n0 + wnL + nj * 16 + r;
            if (col >= N) continue;
#pragma unroll
            for (int v = 0; v < 8; ++v) {
                int row = m0 + wmL + mi * 16 + lh * 8 + v;
                if (row >= M) continue;
                float val = acc[mi][nj][v];
                if (bias) val += bias[col];
                if (Cadd) val += Cadd[bD + (long long)row * ldd + col];
                if (act == 1)      val = fmaxf(val, 0.f);
                else if (act == 2) val = tanhf(val);
                D[bD + (long long)row * ldd + col] = val;
            }
        }
    }
}

// ---------------------------------------------------------------------------
// Elementwise / small kernels
// ---------------------------------------------------------------------------
__global__ void zero_fill_k(float* p, int n) {
    int i = blockIdx.x * blockDim.x + threadIdx.x;
    if (i < n) p[i] = 0.f;
}

// E[b,l,:] = emb[(int)x[b,l], :]
__global__ void embed_gather_k(const float* __restrict__ x,
                               const float* __restrict__ emb,
                               float* __restrict__ E, int total) {
    int i = blockIdx.x * blockDim.x + threadIdx.x;
    if (i >= total) return;
    int c = i & (EMBD - 1);
    int m = i >> 8;                 // EMBD==256
    int b = m / SEQ, l = m - b * SEQ;
    int id = (int)x[(size_t)b * XCOLS + l];
    id = id < 0 ? 0 : (id > 29 ? 29 : id);
    E[i] = emb[(size_t)id * EMBD + c];
}

// h[:,124:128] = aux
__global__ void aux_fill_k(const float* __restrict__ x, float* __restrict__ h, int total) {
    int i = blockIdx.x * blockDim.x + threadIdx.x;
    if (i >= total) return;
    int j = i & 3;
    int m = i >> 2;
    int b = m / SEQ;
    h[(size_t)m * DIN + 124 + j] = x[(size_t)b * XCOLS + 70 + j];
}

// Xp (B,72,128): row lp=0/71 zero, else h[b, lp-1, :]
__global__ void pad_copy_k(const float* __restrict__ h, float* __restrict__ Xp, int total) {
    int i = blockIdx.x * blockDim.x + threadIdx.x;
    if (i >= total) return;
    int c  = i & (DIN - 1);
    int rr = i >> 7;
    int b  = rr / 72, lp = rr - b * 72;
    Xp[i] = (lp >= 1 && lp <= SEQ) ? h[((size_t)b * SEQ + lp - 1) * DIN + c] : 0.f;
}

// zero rows lp=0 and lp=71 of a padded (B,72,128) buffer
__global__ void zero_pad_rows_k(float* __restrict__ Y, int total) {
    int i = blockIdx.x * blockDim.x + threadIdx.x;
    if (i >= total) return;
    int c = i & (DIN - 1);
    int t = i >> 7;
    int b = t >> 1;
    int lp = (t & 1) ? 71 : 0;
    Y[((size_t)b * 72 + lp) * DIN + c] = 0.f;
}

// x2 = h1 + Y2p[:,1:71,:]
__global__ void add_conv_k(const float* __restrict__ h1, const float* __restrict__ Y2p,
                           float* __restrict__ x2, int total) {
    int i = blockIdx.x * blockDim.x + threadIdx.x;
    if (i >= total) return;
    int c = i & (DIN - 1);
    int m = i >> 7;
    int b = m / SEQ, l = m - b * SEQ;
    x2[i] = h1[i] + Y2p[((size_t)b * 72 + l + 1) * DIN + c];
}

// LSTM cell pointwise: gates (B,4H) -> update h/c state, write h into out
__global__ void lstm_point_k(const float* __restrict__ gates,
                             float* __restrict__ hstate, float* __restrict__ cstate,
                             float* __restrict__ out, long long outBatchStride,
                             int H, int nB) {
    int i = blockIdx.x * blockDim.x + threadIdx.x;
    if (i >= nB * H) return;
    int b = i / H, j = i - b * H;
    const float* g = gates + (size_t)b * 4 * H;
    float ig = sigf(g[j]);
    float fg = sigf(g[H + j]);
    float gg = tanhf(g[2 * H + j]);
    float og = sigf(g[3 * H + j]);
    float c = fg * cstate[i] + ig * gg;
    float hv = og * tanhf(c);
    cstate[i] = c;
    hstate[i] = hv;
    out[(long long)b * outBatchStride + j] = hv;
}

// softmax over last dim (contiguous), logits pre-scaled
__global__ void softmax_rows_k(float* __restrict__ S, int rows, int n, float scale) {
    int rI = blockIdx.x * blockDim.x + threadIdx.x;
    if (rI >= rows) return;
    float* p = S + (size_t)rI * n;
    float mx = -1e30f;
    for (int i = 0; i < n; ++i) mx = fmaxf(mx, p[i] * scale);
    float sum = 0.f;
    for (int i = 0; i < n; ++i) { float e = __expf(p[i] * scale - mx); p[i] = e; sum += e; }
    float inv = 1.f / sum;
    for (int i = 0; i < n; ++i) p[i] *= inv;
}

// softmax over axis=1 of (B, Ln, F), in place
__global__ void softmax_axis1_k(float* __restrict__ X, int total, int Ln, int F) {
    int i = blockIdx.x * blockDim.x + threadIdx.x;
    if (i >= total) return;
    int b = i / F, s = i - b * F;
    float* p = X + (size_t)b * Ln * F + s;
    float mx = -1e30f;
    for (int l = 0; l < Ln; ++l) mx = fmaxf(mx, p[(size_t)l * F]);
    float sum = 0.f;
    for (int l = 0; l < Ln; ++l) { float e = __expf(p[(size_t)l * F] - mx); p[(size_t)l * F] = e; sum += e; }
    float inv = 1.f / sum;
    for (int l = 0; l < Ln; ++l) p[(size_t)l * F] *= inv;
}

// L2-normalize rows in place
__global__ void rownorm_k(float* __restrict__ X, int rows, int n) {
    int rI = blockIdx.x * blockDim.x + threadIdx.x;
    if (rI >= rows) return;
    float* p = X + (size_t)rI * n;
    float s = 0.f;
    for (int i = 0; i < n; ++i) s += p[i] * p[i];
    float inv = 1.f / fmaxf(sqrtf(s), 1e-12f);
    for (int i = 0; i < n; ++i) p[i] *= inv;
}

// L2-normalize columns of (k x n) into Wdst
__global__ void colnorm_k(const float* __restrict__ Wsrc, float* __restrict__ Wdst,
                          int k, int n) {
    int c = blockIdx.x * blockDim.x + threadIdx.x;
    if (c >= n) return;
    float s = 0.f;
    for (int i = 0; i < k; ++i) { float v = Wsrc[(size_t)i * n + c]; s += v * v; }
    float inv = 1.f / fmaxf(sqrtf(s), 1e-12f);
    for (int i = 0; i < k; ++i) Wdst[(size_t)i * n + c] = Wsrc[(size_t)i * n + c] * inv;
}

// ---------------------------------------------------------------------------
// Host-side helpers
// ---------------------------------------------------------------------------
static inline dim3 ew(long long n) { return dim3((unsigned)((n + 255) / 256)); }

static inline void launch_gemm(hipStream_t st, const float* A, const float* B,
                               const float* bias, const float* Cadd, float* D,
                               int M, int N, int K, int lda, int ldb, int ldd,
                               long long bsA, long long bsB, long long bsD, int batch,
                               int transA, int transB, int act) {
    dim3 grid((N + BN - 1) / BN, (M + BM - 1) / BM, batch);
    gemm_wmma_f16<<<grid, TPB, 0, st>>>(A, B, bias, Cadd, D, M, N, K,
                                        lda, ldb, ldd, bsA, bsB, bsD,
                                        transA, transB, act);
}

// One LSTM direction: per step t, gates = x_t @ Wih^T + b + h @ Whh^T, then cell.
static void run_lstm_dir(hipStream_t st, const float* x, int Din,
                         const float* Wih, const float* Whh, const float* bv,
                         int H, float* out, int outLd, int colOff, bool backward,
                         float* gates, float* hs, float* cstate) {
    zero_fill_k<<<ew((long long)BATCH * H), 256, 0, st>>>(hs, BATCH * H);
    zero_fill_k<<<ew((long long)BATCH * H), 256, 0, st>>>(cstate, BATCH * H);
    for (int s = 0; s < SEQ; ++s) {
        int t = backward ? (SEQ - 1 - s) : s;
        // gates = x_t @ Wih^T + b
        launch_gemm(st, x + (long long)t * Din, Wih, bv, nullptr, gates,
                    BATCH, 4 * H, Din, SEQ * Din, Din, 4 * H, 0, 0, 0, 1, 0, 1, 0);
        // gates += h @ Whh^T
        launch_gemm(st, hs, Whh, nullptr, gates, gates,
                    BATCH, 4 * H, H, H, H, 4 * H, 0, 0, 0, 1, 0, 1, 0);
        lstm_point_k<<<ew((long long)BATCH * H), 256, 0, st>>>(
            gates, hs, cstate, out + (long long)t * outLd + colOff,
            (long long)SEQ * outLd, H, BATCH);
    }
}

// ---------------------------------------------------------------------------
extern "C" void kernel_launch(void* const* d_in, const int* in_sizes, int n_in,
                              void* d_out, int out_size, void* d_ws, size_t ws_size,
                              hipStream_t stream) {
    (void)in_sizes; (void)out_size;
    if (n_in < 32) return;

    const float* x        = (const float*)d_in[0];
    const float* emb      = (const float*)d_in[1];
    const float* lin_w    = (const float*)d_in[2];
    const float* lin_b    = (const float*)d_in[3];
    const float* lanF_Wih = (const float*)d_in[4];
    const float* lanF_Whh = (const float*)d_in[5];
    const float* lanF_b   = (const float*)d_in[6];
    const float* lanB_Wih = (const float*)d_in[7];
    const float* lanB_Whh = (const float*)d_in[8];
    const float* lanB_b   = (const float*)d_in[9];
    const float* cnn0_w   = (const float*)d_in[10];
    const float* cnn0_b   = (const float*)d_in[11];
    const float* cnn1_w   = (const float*)d_in[12];
    const float* cnn1_b   = (const float*)d_in[13];
    const float* l2F_Wih  = (const float*)d_in[14];
    const float* l2F_Whh  = (const float*)d_in[15];
    const float* l2F_b    = (const float*)d_in[16];
    const float* l2B_Wih  = (const float*)d_in[17];
    const float* l2B_Whh  = (const float*)d_in[18];
    const float* l2B_b    = (const float*)d_in[19];
    const float* fcn1_w   = (const float*)d_in[20];
    const float* fcn1_b   = (const float*)d_in[21];
    const float* fcn2_w   = (const float*)d_in[22];
    const float* fcn2_b   = (const float*)d_in[23];
    const float* fcn2b_w  = (const float*)d_in[24];
    const float* fcn2b_b  = (const float*)d_in[25];
    const float* msa_w    = (const float*)d_in[26];
    const float* msa_b    = (const float*)d_in[27];
    const float* msa_nw   = (const float*)d_in[28];
    const float* fcn4_w   = (const float*)d_in[29];
    const float* fcn4_b   = (const float*)d_in[30];
    const float* fcn5_nw  = (const float*)d_in[31];

    // ---- workspace layout (floats), with two overlays ----
    float* W = (float*)d_ws;
    size_t p = 0;
    float* E    = W + p; p += (size_t)BATCH * SEQ * EMBD;          // reused as pooled
    float* h    = W + p; p += (size_t)BATCH * SEQ * DIN;
    float* XpB  = W + p; p += (size_t)(BATCH * 72 + 2) * DIN;
    float* Xp   = XpB + DIN;                                       // slack row front/back
    size_t mark = p;
    float* hlb  = W + p; p += (size_t)BATCH * SEQ * DIN;
    float* S    = W + p; p += (size_t)BATCH * SEQ * SEQ;
    float* h1   = W + p; p += (size_t)BATCH * SEQ * DIN;
    float* Y1B  = W + p; p += (size_t)(BATCH * 72 + 2) * DIN;
    float* Y1p  = Y1B + DIN;
    float* Y2p  = W + p; p += (size_t)BATCH * 72 * DIN;
    float* hh   = W + mark;   // overlays hlb..Y2p (84.5M floats >= 73.4M needed)
    float* x2   = W + p; p += (size_t)BATCH * SEQ * DIN;
    float* u    = W + p; p += (size_t)BATCH * SEQ * 144;
    float* u2   = W + p; p += (size_t)BATCH * SEQ * 32;
    float* m1   = W + p; p += (size_t)BATCH * 256;
    float* msao = W + p; p += (size_t)BATCH * 64;
    float* Wn   = W + p; p += (size_t)256 * 64;
    float* Wn5  = W + p; p += (size_t)256 * 8;
    float* obuf = W + p; p += (size_t)BATCH * 256;
    float* gates= W + p; p += (size_t)BATCH * 1024;
    float* hs   = W + p; p += (size_t)BATCH * 256;
    float* cst  = W + p; p += (size_t)BATCH * 256;
    float* pooled = E;                                             // overlay (E is dead)
    if (ws_size < p * sizeof(float)) return;

    float* out0   = (float*)d_out;                 // (B,6)
    float* out_cs = out0 + (size_t)BATCH * 6;      // (B,70,11)

    const int BL = BATCH * SEQ;

    // 1) embedding + projection + aux concat -> h (B,70,128)
    embed_gather_k<<<ew((long long)BL * EMBD), 256, 0, stream>>>(x, emb, E, BL * EMBD);
    launch_gemm(stream, E, lin_w, lin_b, nullptr, h,
                BL, 124, EMBD, EMBD, 124, DIN, 0, 0, 0, 1, 0, 0, 1 /*relu*/);
    aux_fill_k<<<ew((long long)BL * 4), 256, 0, stream>>>(x, h, BL * 4);

    // 2) lan BiLSTM (H=64) -> hlb (B,70,128)
    run_lstm_dir(stream, h, DIN, lanF_Wih, lanF_Whh, lanF_b, 64, hlb, DIN, 0,  false, gates, hs, cst);
    run_lstm_dir(stream, h, DIN, lanB_Wih, lanB_Whh, lanB_b, 64, hlb, DIN, 64, true,  gates, hs, cst);

    // 3) self-attention: S = softmax(hl hl^T / sqrt(128)); h1 = S @ hl
    launch_gemm(stream, hlb, hlb, nullptr, nullptr, S,
                SEQ, SEQ, DIN, DIN, DIN, SEQ,
                (long long)SEQ * DIN, (long long)SEQ * DIN, (long long)SEQ * SEQ,
                BATCH, 0, 1, 0);
    softmax_rows_k<<<ew((long long)BL), 256, 0, stream>>>(S, BL, SEQ, 0.08838834764831845f);
    launch_gemm(stream, S, hlb, nullptr, nullptr, h1,
                SEQ, DIN, SEQ, SEQ, DIN, DIN,
                (long long)SEQ * SEQ, (long long)SEQ * DIN, (long long)SEQ * DIN,
                BATCH, 0, 0, 0);

    // 4) two conv1d(k=3, SAME, relu) as 3 shifted GEMMs over padded (B,72,128)
    pad_copy_k<<<ew((long long)BATCH * 72 * DIN), 256, 0, stream>>>(h, Xp, BATCH * 72 * DIN);
    for (int k = 0; k < 3; ++k)
        launch_gemm(stream, Xp + (long long)(k - 1) * DIN, cnn0_w + (size_t)k * DIN * DIN,
                    (k == 2) ? cnn0_b : nullptr, (k == 0) ? nullptr : Y1p, Y1p,
                    BATCH * 72, DIN, DIN, DIN, DIN, DIN, 0, 0, 0, 1, 0, 0, (k == 2) ? 1 : 0);
    zero_pad_rows_k<<<ew((long long)BATCH * 2 * DIN), 256, 0, stream>>>(Y1p, BATCH * 2 * DIN);
    for (int k = 0; k < 3; ++k)
        launch_gemm(stream, Y1p + (long long)(k - 1) * DIN, cnn1_w + (size_t)k * DIN * DIN,
                    (k == 2) ? cnn1_b : nullptr, (k == 0) ? nullptr : Y2p, Y2p,
                    BATCH * 72, DIN, DIN, DIN, DIN, DIN, 0, 0, 0, 1, 0, 0, (k == 2) ? 1 : 0);
    add_conv_k<<<ew((long long)BL * DIN), 256, 0, stream>>>(h1, Y2p, x2, BL * DIN);

    // 5) lstm2 BiLSTM (H=256) on x2 -> hh (B,70,512)   [hh overlays dead buffers]
    run_lstm_dir(stream, x2, DIN, l2F_Wih, l2F_Whh, l2F_b, 256, hh, 512, 0,   false, gates, hs, cst);
    run_lstm_dir(stream, x2, DIN, l2B_Wih, l2B_Whh, l2B_b, 256, hh, 512, 256, true,  gates, hs, cst);

    // 6) heads
    launch_gemm(stream, hh, fcn1_w, fcn1_b, nullptr, u,
                BL, 144, 512, 512, 144, 144, 0, 0, 0, 1, 0, 0, 2 /*tanh*/);
    launch_gemm(stream, u, fcn2_w, fcn2_b, nullptr, u2,
                BL, 32, 144, 144, 32, 32, 0, 0, 0, 1, 0, 0, 0);
    launch_gemm(stream, u2, fcn2b_w, fcn2b_b, nullptr, out_cs,
                BL, 11, 32, 32, 11, 11, 0, 0, 0, 1, 0, 0, 0);

    // a = softmax(hh, axis=1) in place; pooled[b] = a^T @ u2  (512x32)
    softmax_axis1_k<<<ew((long long)BATCH * 512), 256, 0, stream>>>(hh, BATCH * 512, SEQ, 512);
    launch_gemm(stream, hh, u2, nullptr, nullptr, pooled,
                512, 32, SEQ, 512, 32, 32,
                (long long)SEQ * 512, (long long)SEQ * 32, (long long)512 * 32,
                BATCH, 1 /*transA*/, 0, 0);

    // msa branch: m1 = msa_feat @ msa_w + b; normed linear with msa_nw
    launch_gemm(stream, x + 74, msa_w, msa_b, nullptr, m1,
                BATCH, 256, 768, XCOLS, 256, 256, 0, 0, 0, 1, 0, 0, 0);
    rownorm_k<<<ew((long long)BATCH), 256, 0, stream>>>(m1, BATCH, 256);
    colnorm_k<<<ew(64), 256, 0, stream>>>(msa_nw, Wn, 256, 64);
    launch_gemm(stream, m1, Wn, nullptr, nullptr, msao,
                BATCH, 64, 256, 256, 64, 64, 0, 0, 0, 1, 0, 0, 0);

    // fcn4: o = relu( pooled @ W4[:16384] + msa @ W4[16384:] + b )
    launch_gemm(stream, pooled, fcn4_w, nullptr, nullptr, obuf,
                BATCH, 256, 16384, 16384, 256, 256, 0, 0, 0, 1, 0, 0, 0);
    launch_gemm(stream, msao, fcn4_w + (size_t)16384 * 256, fcn4_b, obuf, obuf,
                BATCH, 256, 64, 64, 256, 256, 0, 0, 0, 1, 0, 0, 1 /*relu*/);

    // final normed linear -> out0 (B,6)
    rownorm_k<<<ew((long long)BATCH), 256, 0, stream>>>(obuf, BATCH, 256);
    colnorm_k<<<ew(6), 256, 0, stream>>>(fcn5_nw, Wn5, 256, 6);
    launch_gemm(stream, obuf, Wn5, nullptr, nullptr, out0,
                BATCH, 6, 256, 256, 6, 6, 0, 0, 0, 1, 0, 0, 0);
}
